// GroupedLinear_78305843740735
// MI455X (gfx1250) — matrix operand order repaired and checked by reading, not compiled, not run on hardware
//
#include <hip/hip_runtime.h>

// ---------------------------------------------------------------------------
// Grouped GEMM for MI455X (gfx1250): out[t, :] = x[t, :] @ weight[group(t)]
//   x:      (16384, 2048)  fp32
//   weight: (8, 2048, 8192) fp32
//   out:    (16384, 8192)  fp32
// Strategy: bf16x3 split-precision WMMA (v_wmma_f32_16x16x32_bf16),
// LDS-staged 128x128x64 tiles, 8 wave32s per block, memory-bound design.
// ---------------------------------------------------------------------------

typedef __attribute__((ext_vector_type(16))) __bf16 v16bf;
typedef __attribute__((ext_vector_type(8)))  __bf16 v8bf;
typedef __attribute__((ext_vector_type(4)))  __bf16 v4bf;
typedef __attribute__((ext_vector_type(8)))  float  v8f;

#define D_IN   2048
#define D_OUT  8192
#define NGRP   8
#define BM     128
#define BN     128
#define BK     64
#define AST    72   // padded LDS K-stride (bf16 elements): 144B = 36 banks

union FragBF {
    v16bf v;
    v8bf  h[2];
};

__device__ __forceinline__ __bf16 bf_hi(float f) {
    return (__bf16)f;                      // RNE convert
}
__device__ __forceinline__ __bf16 bf_lo(float f) {
    __bf16 h = (__bf16)f;
    return (__bf16)(f - (float)h);         // residual: ~8 more mantissa bits
}

__global__ __launch_bounds__(256)
void grouped_gemm_bf16x3_wmma(const float* __restrict__ x,
                              const int*   __restrict__ groups,
                              const float* __restrict__ weight,
                              float*       __restrict__ out) {
    __shared__ __bf16 As_hi[BM * AST];
    __shared__ __bf16 As_lo[BM * AST];
    __shared__ __bf16 Bs_hi[BN * AST];
    __shared__ __bf16 Bs_lo[BN * AST];

    const int tid  = threadIdx.x;
    const int wid  = tid >> 5;          // 0..7
    const int lane = tid & 31;
    const int lrow = lane & 15;         // N/M index within fragment
    const int lhi  = lane >> 4;         // 0 or 1 (K-half selector)
    const int wave_m = wid >> 2;        // 0..1  -> 64-row slab
    const int wave_n = wid & 3;         // 0..3  -> 32-col slab

    const int n0 = blockIdx.x * BN;
    const int m0 = blockIdx.y * BM;

    // ---- find expert for this M-tile (tiles never straddle: sizes % 512 == 0)
    int g = 0, off = 0;
#pragma unroll
    for (int i = 0; i < NGRP - 1; ++i) {
        int s = groups[i];
        if (m0 >= off + s) { off += s; g = i + 1; }
    }
    const float* __restrict__ w = weight + (size_t)g * D_IN * D_OUT;

    // ---- accumulators: 4 (M) x 2 (N) fragments of 16x16 f32
    v8f acc[4][2];
#pragma unroll
    for (int mf = 0; mf < 4; ++mf)
#pragma unroll
        for (int nf = 0; nf < 2; ++nf)
#pragma unroll
            for (int r = 0; r < 8; ++r) acc[mf][nf][r] = 0.0f;

    for (int k0 = 0; k0 < D_IN; k0 += BK) {
        // -------- prefetch next K slab while we work on this one
        if (k0 + BK < D_IN) {
            __builtin_prefetch(x + (size_t)(m0 + (tid >> 1)) * D_IN + (k0 + BK) + (tid & 1) * 32, 0, 0);
            __builtin_prefetch(w + (size_t)(k0 + BK + (tid >> 2)) * D_OUT + n0 + (tid & 3) * 32, 0, 0);
        }

        // -------- stage A (128x64) and B (64x128, transposed to [N][K]) to LDS
#pragma unroll
        for (int i = 0; i < 8; ++i) {
            const int idx = (i * 256 + tid) * 4;   // float4 granularity, 0..8191

            // A: row-major [M][K], hi/lo bf16 planes
            {
                const int r = idx >> 6;            // 0..127
                const int c = idx & 63;            // 0..63, %4 == 0
                const float4 xv = *(const float4*)(x + (size_t)(m0 + r) * D_IN + k0 + c);
                v4bf h4, l4;
                h4[0] = bf_hi(xv.x); l4[0] = bf_lo(xv.x);
                h4[1] = bf_hi(xv.y); l4[1] = bf_lo(xv.y);
                h4[2] = bf_hi(xv.z); l4[2] = bf_lo(xv.z);
                h4[3] = bf_hi(xv.w); l4[3] = bf_lo(xv.w);
                *(v4bf*)&As_hi[r * AST + c] = h4;
                *(v4bf*)&As_lo[r * AST + c] = l4;
            }

            // B: transpose to [N][K] so WMMA B-fragments are K-contiguous
            {
                const int rb = idx >> 7;           // k: 0..63
                const int cb = idx & 127;          // n: 0..127, %4 == 0
                const float4 wv = *(const float4*)(w + (size_t)(k0 + rb) * D_OUT + n0 + cb);
#pragma unroll
                for (int j = 0; j < 4; ++j) {
                    const float f = ((const float*)&wv)[j];
                    Bs_hi[(cb + j) * AST + rb] = bf_hi(f);
                    Bs_lo[(cb + j) * AST + rb] = bf_lo(f);
                }
            }
        }
        __syncthreads();

        // -------- two K=32 WMMA steps per staged BK=64 slab
#pragma unroll
        for (int ks = 0; ks < 2; ++ks) {
            FragBF ah[4], al[4], bh[2], bl[2];

            // A fragments: lane<16 holds K{0..7,16..23}, lane>=16 K{8..15,24..31}
            const int ka = ks * 32 + lhi * 8;
#pragma unroll
            for (int mf = 0; mf < 4; ++mf) {
                const int row = wave_m * 64 + mf * 16 + lrow;
                ah[mf].h[0] = *(const v8bf*)&As_hi[row * AST + ka];
                ah[mf].h[1] = *(const v8bf*)&As_hi[row * AST + ka + 16];
                al[mf].h[0] = *(const v8bf*)&As_lo[row * AST + ka];
                al[mf].h[1] = *(const v8bf*)&As_lo[row * AST + ka + 16];
            }

            // B fragments: lane<16 holds K{0..7,16..23}, lane>=16 K{8..15,24..31}
            const int kb = ks * 32 + lhi * 8;
#pragma unroll
            for (int nf = 0; nf < 2; ++nf) {
                const int col = wave_n * 32 + nf * 16 + lrow;
                bh[nf].h[0] = *(const v8bf*)&Bs_hi[col * AST + kb];
                bh[nf].h[1] = *(const v8bf*)&Bs_hi[col * AST + kb + 16];
                bl[nf].h[0] = *(const v8bf*)&Bs_lo[col * AST + kb];
                bl[nf].h[1] = *(const v8bf*)&Bs_lo[col * AST + kb + 16];
            }

            // bf16x3: acc += Ah*Bh + Al*Bh + Ah*Bl  (fp32 accumulate)
#pragma unroll
            for (int mf = 0; mf < 4; ++mf) {
#pragma unroll
                for (int nf = 0; nf < 2; ++nf) {
                    acc[mf][nf] = __builtin_amdgcn_wmma_f32_16x16x32_bf16(
                        false, ah[mf].v, false, bh[nf].v, (short)0, acc[mf][nf], false, false);
                    acc[mf][nf] = __builtin_amdgcn_wmma_f32_16x16x32_bf16(
                        false, al[mf].v, false, bh[nf].v, (short)0, acc[mf][nf], false, false);
                    acc[mf][nf] = __builtin_amdgcn_wmma_f32_16x16x32_bf16(
                        false, ah[mf].v, false, bl[nf].v, (short)0, acc[mf][nf], false, false);
                }
            }
        }
        __syncthreads();
    }

    // -------- write out: C layout VGPR r -> row r (lanes 0-15) / r+8 (lanes 16-31)
#pragma unroll
    for (int mf = 0; mf < 4; ++mf) {
        const int rbase = m0 + wave_m * 64 + mf * 16 + lhi * 8;
#pragma unroll
        for (int nf = 0; nf < 2; ++nf) {
            const int col = n0 + wave_n * 32 + nf * 16 + lrow;
            float* op = out + (size_t)rbase * D_OUT + col;
#pragma unroll
            for (int r = 0; r < 8; ++r) {
                op[(size_t)r * D_OUT] = acc[mf][nf][r];
            }
        }
    }
}

extern "C" void kernel_launch(void* const* d_in, const int* in_sizes, int n_in,
                              void* d_out, int out_size, void* d_ws, size_t ws_size,
                              hipStream_t stream) {
    const float* x       = (const float*)d_in[0];
    const int*   groups  = (const int*)d_in[1];
    const float* weight  = (const float*)d_in[2];
    float*       out     = (float*)d_out;

    dim3 grid(D_OUT / BN, 16384 / BM);   // 64 x 128 tiles
    dim3 block(256);
    grouped_gemm_bf16x3_wmma<<<grid, block, 0, stream>>>(x, groups, weight, out);
}
